// MIXSelector_37538014167384
// MI455X (gfx1250) — compile-verified, matrix operand order checked by reading, not run on hardware
//
#include <hip/hip_runtime.h>
#include <math.h>

// ---------------------------------------------------------------------------
// GAT + LayerNorm + GCN pipeline for gfx1250 (MI455X).
// Dense GEMM h = x @ W_gat runs on v_wmma_f32_16x16x32_f16 with a split-f16
// (hi/lo) 3-WMMA scheme for fp32-class accuracy, using three independent
// accumulators so the WMMAs pipeline instead of RAW-chaining. Edge phases are
// scatter kernels that stay resident in the 192MB L2.
// ---------------------------------------------------------------------------

typedef __attribute__((ext_vector_type(16))) _Float16 v16h;
typedef __attribute__((ext_vector_type(8)))  float    v8f;

#define IN_CH 256
#define HEADS 4
#define CPH   64
#define HC    256
#define NEG_SLOPE 0.2f
#define LN_EPS 1e-5f

// ---------------------------------------------------------------------------
// K1: h[N,256] = x[N,256] @ W[256,256]  via WMMA f16-split (3 WMMAs / k-step)
// One wave computes one 16x16 output tile. 8 waves / block.
// N is a multiple of 16 and K=256 a multiple of 32, so no predication
// (WMMA requires EXEC all-ones).
// ---------------------------------------------------------------------------
__global__ __launch_bounds__(256) void k_gemm_wmma(const float* __restrict__ x,
                                                   const float* __restrict__ W,
                                                   float* __restrict__ hout) {
  const int wave = threadIdx.x >> 5;
  const int lane = threadIdx.x & 31;
  const int tile = blockIdx.x * 8 + wave;   // 0 .. (N/16)*16 - 1
  const int tm = tile >> 4;                 // M tile (N/16 of them)
  const int tn = tile & 15;                 // N tile (HC/16 = 16 of them)
  const int r16  = lane & 15;
  const int half = lane >> 4;               // 0 or 1 (K-group select)
  const int mrow = tm * 16 + r16;
  const int ncol = tn * 16 + r16;

  // Independent accumulators: xh*Wh, xl*Wh, xh*Wl. Each WMMA only depends on
  // the same accumulator 3 WMMAs back, so the matrix pipe stays busy.
  v8f acc_hh = {};
  v8f acc_lh = {};
  v8f acc_hl = {};
  for (int kt = 0; kt < 8; ++kt) {
    const int k0 = kt * 32;
    v16h a_hi, a_lo, b_hi, b_lo;
#pragma unroll
    for (int j = 0; j < 8; ++j) {
      // ISA 16-bit A layout: VGPR j<4 -> K = half*8 + 2j (+1)
      //                      VGPR j>=4 -> K = 16 + half*8 + 2(j-4) (+1)
      const int k = k0 + ((j < 4) ? (half * 8 + 2 * j)
                                  : (16 + half * 8 + 2 * (j - 4)));
      // A tile: row mrow, cols k, k+1 (row-major x)
      const float ax0 = x[(size_t)mrow * IN_CH + k];
      const float ax1 = x[(size_t)mrow * IN_CH + k + 1];
      const _Float16 ah0 = (_Float16)ax0, ah1 = (_Float16)ax1;
      a_hi[2 * j]     = ah0;
      a_hi[2 * j + 1] = ah1;
      a_lo[2 * j]     = (_Float16)(ax0 - (float)ah0);
      a_lo[2 * j + 1] = (_Float16)(ax1 - (float)ah1);
      // B tile: rows k,k+1 of column ncol (row-major W: [K, HC])
      const float bx0 = W[(size_t)k * HC + ncol];
      const float bx1 = W[(size_t)(k + 1) * HC + ncol];
      const _Float16 bh0 = (_Float16)bx0, bh1 = (_Float16)bx1;
      b_hi[2 * j]     = bh0;
      b_hi[2 * j + 1] = bh1;
      b_lo[2 * j]     = (_Float16)(bx0 - (float)bh0);
      b_lo[2 * j + 1] = (_Float16)(bx1 - (float)bh1);
    }
    acc_hh = __builtin_amdgcn_wmma_f32_16x16x32_f16(false, a_hi, false, b_hi,
                                                    (short)0, acc_hh, false, false);
    acc_lh = __builtin_amdgcn_wmma_f32_16x16x32_f16(false, a_lo, false, b_hi,
                                                    (short)0, acc_lh, false, false);
    acc_hl = __builtin_amdgcn_wmma_f32_16x16x32_f16(false, a_hi, false, b_lo,
                                                    (short)0, acc_hl, false, false);
  }
  // C/D layout: lane = column (n = lane%16), VGPR r -> row r + 8*(lane>=16)
#pragma unroll
  for (int r = 0; r < 8; ++r) {
    const int m = tm * 16 + r + 8 * half;
    hout[(size_t)m * HC + ncol] = acc_hh[r] + (acc_lh[r] + acc_hl[r]);
  }
}

// ---------------------------------------------------------------------------
// K2: attention logits per (node, head): al_s/al_d = <h[n,h,:], a_src/dst[h,:]>
// ---------------------------------------------------------------------------
__global__ void k_node_al(const float* __restrict__ h,
                          const float* __restrict__ a_src,
                          const float* __restrict__ a_dst,
                          float* __restrict__ al_s, float* __restrict__ al_d,
                          int n_nodes) {
  const int i = blockIdx.x * blockDim.x + threadIdx.x;  // node*H + head
  if (i >= n_nodes * HEADS) return;
  const int node = i >> 2, hd = i & 3;
  const float* hp = h + (size_t)node * HC + hd * CPH;
  const float* as = a_src + hd * CPH;
  const float* ad = a_dst + hd * CPH;
  float ss = 0.f, sd = 0.f;
#pragma unroll 8
  for (int c = 0; c < CPH; ++c) {
    const float v = hp[c];
    ss += v * as[c];
    sd += v * ad[c];
  }
  al_s[i] = ss;
  al_d[i] = sd;
}

__device__ __forceinline__ void edge_ends(const int* __restrict__ ei, int e,
                                          int E, int& src, int& dst) {
  if (e < E) { src = ei[e]; dst = ei[E + e]; }
  else       { src = dst = e - E; }          // self loop
}

__device__ __forceinline__ unsigned f32_enc(float f) {
  const unsigned b = __float_as_uint(f);
  return (b & 0x80000000u) ? ~b : (b | 0x80000000u);   // monotonic map
}
__device__ __forceinline__ float f32_dec(unsigned u) {
  const unsigned b = (u & 0x80000000u) ? (u ^ 0x80000000u) : ~u;
  return __uint_as_float(b);
}

// ---------------------------------------------------------------------------
// K3: per (edge, head): e = leaky_relu(al_s[src]+al_d[dst]); segment max +
// degree accumulation. m_enc pre-zeroed (0 == encoded minimum).
// ---------------------------------------------------------------------------
__global__ void k_edge_max_deg(const int* __restrict__ ei,
                               const float* __restrict__ al_s,
                               const float* __restrict__ al_d,
                               unsigned* __restrict__ m_enc,
                               float* __restrict__ deg, int E, int n_nodes) {
  const int t = blockIdx.x * blockDim.x + threadIdx.x;
  const int Et = E + n_nodes;
  if (t >= Et * HEADS) return;
  const int e = t >> 2, hd = t & 3;
  int src, dst;
  edge_ends(ei, e, E, src, dst);
  float ev = al_s[src * HEADS + hd] + al_d[dst * HEADS + hd];
  ev = ev > 0.f ? ev : NEG_SLOPE * ev;
  atomicMax(&m_enc[dst * HEADS + hd], f32_enc(ev));
  if (hd == 0) atomicAdd(&deg[dst], 1.0f);
}

// ---------------------------------------------------------------------------
// K4: per (edge, head): p = exp(e - m[dst]); store p; z[dst] += p.
// ---------------------------------------------------------------------------
__global__ void k_edge_softmax_num(const int* __restrict__ ei,
                                   const float* __restrict__ al_s,
                                   const float* __restrict__ al_d,
                                   const unsigned* __restrict__ m_enc,
                                   float* __restrict__ p, float* __restrict__ z,
                                   int E, int n_nodes) {
  const int t = blockIdx.x * blockDim.x + threadIdx.x;
  const int Et = E + n_nodes;
  if (t >= Et * HEADS) return;
  const int e = t >> 2, hd = t & 3;
  int src, dst;
  edge_ends(ei, e, E, src, dst);
  float ev = al_s[src * HEADS + hd] + al_d[dst * HEADS + hd];
  ev = ev > 0.f ? ev : NEG_SLOPE * ev;
  const float m = f32_dec(m_enc[dst * HEADS + hd]);
  const float pv = __expf(ev - m);
  p[t] = pv;
  atomicAdd(&z[dst * HEADS + hd], pv);
}

// ---------------------------------------------------------------------------
// K5: message aggregation. One 256-thread block per edge; thread = channel.
// out1[dst, t] += h[src, t] * alpha[e, head(t)].  (L2-resident atomics)
// ---------------------------------------------------------------------------
__global__ __launch_bounds__(256) void k_edge_message(
    const int* __restrict__ ei, const float* __restrict__ h,
    const float* __restrict__ p, const float* __restrict__ z,
    float* __restrict__ out1, int E, int n_nodes) {
  const int e = blockIdx.x;
  const int t = threadIdx.x;      // channel 0..255
  const int hd = t >> 6;          // head = channel / 64
  int src, dst;
  edge_ends(ei, e, E, src, dst);
  const float alpha = p[e * HEADS + hd] / z[dst * HEADS + hd];
  atomicAdd(&out1[(size_t)dst * HC + t], h[(size_t)src * HC + t] * alpha);
}

// ---------------------------------------------------------------------------
// K6: per node (1 block / node): bias + ReLU + LayerNorm + dot(W_gcn).
// Also finalizes dinv = rsqrt(deg) and seeds d_out with b_gcn.
// ---------------------------------------------------------------------------
__global__ __launch_bounds__(256) void k_node_finish(
    const float* __restrict__ out1, const float* __restrict__ b_gat,
    const float* __restrict__ gamma, const float* __restrict__ beta,
    const float* __restrict__ Wg, const float* __restrict__ bg,
    const float* __restrict__ deg, float* __restrict__ h2,
    float* __restrict__ dinv, float* __restrict__ out) {
  __shared__ float red[256];
  const int n = blockIdx.x, t = threadIdx.x;
  float v = out1[(size_t)n * HC + t] + b_gat[t];
  v = v > 0.f ? v : 0.f;                       // ReLU

  red[t] = v; __syncthreads();
  for (int s = 128; s > 0; s >>= 1) { if (t < s) red[t] += red[t + s]; __syncthreads(); }
  const float mu = red[0] * (1.0f / HC); __syncthreads();

  const float d = v - mu;
  red[t] = d * d; __syncthreads();
  for (int s = 128; s > 0; s >>= 1) { if (t < s) red[t] += red[t + s]; __syncthreads(); }
  const float var = red[0] * (1.0f / HC); __syncthreads();

  const float a = d * rsqrtf(var + LN_EPS) * gamma[t] + beta[t];

  red[t] = a * Wg[t]; __syncthreads();
  for (int s = 128; s > 0; s >>= 1) { if (t < s) red[t] += red[t + s]; __syncthreads(); }
  if (t == 0) {
    h2[n]   = red[0];
    dinv[n] = rsqrtf(deg[n]);
    out[n]  = bg[0];                            // seed output with bias
  }
}

// ---------------------------------------------------------------------------
// K7: GCN scatter: out[dst] += h2[src] * dinv[src] * dinv[dst]
// ---------------------------------------------------------------------------
__global__ void k_edge_gcn(const int* __restrict__ ei,
                           const float* __restrict__ h2,
                           const float* __restrict__ dinv,
                           float* __restrict__ out, int E, int n_nodes) {
  const int e = blockIdx.x * blockDim.x + threadIdx.x;
  if (e >= E + n_nodes) return;
  int src, dst;
  edge_ends(ei, e, E, src, dst);
  atomicAdd(&out[dst], h2[src] * dinv[src] * dinv[dst]);
}

// ---------------------------------------------------------------------------
extern "C" void kernel_launch(void* const* d_in, const int* in_sizes, int n_in,
                              void* d_out, int out_size, void* d_ws,
                              size_t ws_size, hipStream_t stream) {
  const float* x      = (const float*)d_in[0];
  const int*   ei     = (const int*)d_in[1];
  const float* W_gat  = (const float*)d_in[2];
  const float* a_src  = (const float*)d_in[3];
  const float* a_dst  = (const float*)d_in[4];
  const float* b_gat  = (const float*)d_in[5];
  const float* gamma  = (const float*)d_in[6];
  const float* beta   = (const float*)d_in[7];
  const float* W_gcn  = (const float*)d_in[8];
  const float* b_gcn  = (const float*)d_in[9];
  float* out = (float*)d_out;

  const int N  = in_sizes[0] / IN_CH;   // 10000
  const int E  = in_sizes[1] / 2;       // 320000
  const int Et = E + N;

  // ---- workspace carve-out ----
  char* base = (char*)d_ws;
  const size_t szH = (size_t)N * HC * sizeof(float);
  float*    h     = (float*)(base);
  float*    out1  = (float*)(base + szH);
  float*    al_s  = (float*)(base + 2 * szH);
  float*    al_d  = al_s + (size_t)N * HEADS;
  unsigned* m_enc = (unsigned*)(al_d + (size_t)N * HEADS);
  float*    z     = (float*)(m_enc + (size_t)N * HEADS);
  float*    deg   = z + (size_t)N * HEADS;
  float*    h2    = deg + N;
  float*    dinv  = h2 + N;
  float*    p     = dinv + N;           // Et*HEADS floats

  // ---- zero accumulators (capture-safe async memsets) ----
  hipMemsetAsync(out1,  0, szH, stream);
  hipMemsetAsync(m_enc, 0, (size_t)N * HEADS * sizeof(unsigned), stream); // 0 == encoded -min
  hipMemsetAsync(z,     0, (size_t)N * HEADS * sizeof(float), stream);
  hipMemsetAsync(deg,   0, (size_t)N * sizeof(float), stream);

  // K1: GEMM (N/16 * 16 tiles, 8 waves/block)
  const int tiles = (N / 16) * (HC / 16);
  k_gemm_wmma<<<tiles / 8, 256, 0, stream>>>(x, W_gat, h);

  // K2: attention logits
  k_node_al<<<(N * HEADS + 255) / 256, 256, 0, stream>>>(h, a_src, a_dst,
                                                         al_s, al_d, N);
  // K3: segment max + degree
  k_edge_max_deg<<<(Et * HEADS + 255) / 256, 256, 0, stream>>>(ei, al_s, al_d,
                                                               m_enc, deg, E, N);
  // K4: softmax numerators + denominators
  k_edge_softmax_num<<<(Et * HEADS + 255) / 256, 256, 0, stream>>>(
      ei, al_s, al_d, m_enc, p, z, E, N);
  // K5: message aggregation (1 block per edge)
  k_edge_message<<<Et, 256, 0, stream>>>(ei, h, p, z, out1, E, N);
  // K6: ReLU + LayerNorm + GCN projection per node
  k_node_finish<<<N, 256, 0, stream>>>(out1, b_gat, gamma, beta, W_gcn, b_gcn,
                                       deg, h2, dinv, out);
  // K7: GCN normalized scatter
  k_edge_gcn<<<(Et + 255) / 256, 256, 0, stream>>>(ei, h2, dinv, out, E, N);
}
